// RingDilatedAttentionHilbertProper_30331059044983
// MI455X (gfx1250) — matrix operand order repaired and checked by reading, not compiled, not run on hardware
//
#include <hip/hip_runtime.h>
#include <hip/hip_bf16.h>
#include <math.h>

typedef __attribute__((ext_vector_type(16))) __bf16 v16bf;
typedef __attribute__((ext_vector_type(8)))  __bf16 v8bf;
typedef __attribute__((ext_vector_type(8)))  float  v8f;

#define BATCH   2
#define SEQ     8192
#define EMBED   512
#define NHEADS  16
#define HDIM    32
#define QSCALE  0.17677669529663689f   // 1/sqrt(32)

// ---------------------------------------------------------------------------
// Kernel 1: Hilbert group assignment.
// For each segment: key[p] = xy2d(side, p%side, p/side); rank[p] = #{keys < key[p]}
// (brute force in LDS, L <= 4096); group c = rank % r; deterministic slot =
// #{p' < p with same group}; gidx[segBase + c*g + slot] = p.
// ---------------------------------------------------------------------------
__global__ __launch_bounds__(1024) void hilbert_groups_kernel(int* __restrict__ gidx) {
    __shared__ unsigned keys[4096];
    __shared__ unsigned char grp[4096];
    const int Ls[4]    = {1024, 2048, 4096, 1024};
    const int Rs[4]    = {1, 2, 4, 8};
    const int Sides[4] = {32, 64, 64, 32};
    const int Base[4]  = {0, 1024, 3072, 7168};
    const int seg = blockIdx.x;
    const int L = Ls[seg], r = Rs[seg], side = Sides[seg], base = Base[seg];
    const int g = L / r;

    for (int p = threadIdx.x; p < L; p += blockDim.x) {
        unsigned x = (unsigned)(p % side), y = (unsigned)(p / side);
        unsigned d = 0;
        for (unsigned s = (unsigned)side >> 1; s > 0; s >>= 1) {
            unsigned rx = (x & s) ? 1u : 0u;
            unsigned ry = (y & s) ? 1u : 0u;
            d += s * s * ((3u * rx) ^ ry);
            if (ry == 0) {
                if (rx) { x = s - 1 - x; y = s - 1 - y; }
                unsigned t = x; x = y; y = t;
            }
        }
        keys[p] = d;
    }
    __syncthreads();
    for (int p = threadIdx.x; p < L; p += blockDim.x) {
        unsigned kp = keys[p];
        int rank = 0;
        for (int q = 0; q < L; ++q) rank += (keys[q] < kp) ? 1 : 0;
        grp[p] = (unsigned char)(rank % r);
    }
    __syncthreads();
    for (int p = threadIdx.x; p < L; p += blockDim.x) {
        unsigned char c = grp[p];
        int slot = 0;
        for (int q = 0; q < p; ++q) slot += (grp[q] == c) ? 1 : 0;
        gidx[base + (int)c * g + slot] = p;
    }
}

// ---------------------------------------------------------------------------
// Kernel 2: QKV projection GEMM, M=B*S=16384, N=1536, K=512, WMMA bf16.
// Block tile 128x64, 256 threads (8 waves), wave w -> rows [16w,16w+16).
// Epilogue: +bias, q scaled by 1/sqrt(D), bf16 out to [B,H,S,D] tensors.
// ---------------------------------------------------------------------------
__global__ __launch_bounds__(256) void qkv_gemm_kernel(
        const float* __restrict__ x, const float* __restrict__ w,
        const float* __restrict__ bias,
        __bf16* __restrict__ qb, __bf16* __restrict__ kb, __bf16* __restrict__ vb) {
    __shared__ __align__(32) __bf16 Alds[128 * 32];
    __shared__ __align__(32) __bf16 Bt[64 * 32];
    const int tid = threadIdx.x;
    const int lane = tid & 31, wv = tid >> 5;
    const int m0 = blockIdx.x * 128;
    const int n0 = blockIdx.y * 64;
    const int nl = lane & 15, hb = lane >> 4;

    v8f acc[4] = {v8f{}, v8f{}, v8f{}, v8f{}};

    for (int k0 = 0; k0 < EMBED; k0 += 32) {
        {   // stage A: 128x32 f32 -> bf16 LDS (row-major)
            int row = tid >> 1, half = tid & 1;
            const float* src = x + (size_t)(m0 + row) * EMBED + k0 + half * 16;
            __bf16* dst = &Alds[row * 32 + half * 16];
            #pragma unroll
            for (int i = 0; i < 16; ++i) dst[i] = (__bf16)src[i];
        }
        {   // stage B transposed: w[k][n] -> Bt[n][k]
            int k = tid >> 3, ns = (tid & 7) * 8;
            const float* src = w + (size_t)(k0 + k) * 1536 + n0 + ns;
            #pragma unroll
            for (int i = 0; i < 8; ++i) Bt[(ns + i) * 32 + k] = (__bf16)src[i];
        }
        __syncthreads();
        // A operand: lane<16 row nl gets K {0..15}; lane>=16 gets {16..31}
        v8bf alo = *(const v8bf*)&Alds[(wv * 16 + nl) * 32 + hb * 16];
        v8bf ahi = *(const v8bf*)&Alds[(wv * 16 + nl) * 32 + hb * 16 + 8];
        v16bf a;
        #pragma unroll
        for (int i = 0; i < 8; ++i) { a[i] = alo[i]; a[8 + i] = ahi[i]; }
        #pragma unroll
        for (int j = 0; j < 4; ++j) {
            v16bf bmat = *(const v16bf*)&Bt[(j * 16 + nl) * 32 + hb * 16];
            acc[j] = __builtin_amdgcn_wmma_f32_16x16x32_bf16(
                false, a, false, bmat, (short)0, acc[j], false, false);
        }
        __syncthreads();
    }
    // Epilogue: C layout lane l holds (m = i + 8*(l>>4), n = l&15) in acc[j][i]
    #pragma unroll
    for (int j = 0; j < 4; ++j) {
        int n = n0 + j * 16 + nl;
        float bval = bias[n];
        int which = n >> 9;
        int hd = n & 511;
        int h = hd >> 5, d = hd & 31;
        __bf16* dst = (which == 0) ? qb : (which == 1) ? kb : vb;
        float scale = (which == 0) ? QSCALE : 1.0f;
        #pragma unroll
        for (int i = 0; i < 8; ++i) {
            int row = m0 + wv * 16 + i + 8 * hb;
            int bb = row >> 13, s = row & (SEQ - 1);
            float val = (acc[j][i] + bval) * scale;
            dst[((size_t)(bb * NHEADS + h) * SEQ + s) * HDIM + d] = (__bf16)val;
        }
    }
}

// ---------------------------------------------------------------------------
// Kernel 3: dilated attention over gathered groups (flash-style online softmax).
// 800 blocks x 256 threads. Block -> (seg, b, h, qtile of 128 rows); wave ->
// 16-row Q tile. Loop over keys in 32-chunks: K row-major + V transposed in
// LDS; 2 score WMMAs + 2 PV WMMAs per chunk. Output scattered bf16 by gidx.
// ---------------------------------------------------------------------------
__global__ __launch_bounds__(256) void attn_kernel(
        const __bf16* __restrict__ qb, const __bf16* __restrict__ kb,
        const __bf16* __restrict__ vb, const int* __restrict__ gidx,
        __bf16* __restrict__ ob) {
    __shared__ __align__(32) __bf16 Klds[32 * 32];
    __shared__ __align__(32) __bf16 Vt[32 * 32];
    __shared__ __align__(32) __bf16 Plds[8][16 * 32];

    const int segStart[4] = {0, 1024, 3072, 7168};
    const int segR[4]     = {1, 2, 4, 8};
    const int segG[4]     = {1024, 1024, 1024, 128};

    int bid = blockIdx.x;
    int seg, bh, tile;
    if (bid < 768) { seg = bid >> 8; int rem = bid & 255; bh = rem >> 3; tile = rem & 7; }
    else           { seg = 3; bh = bid - 768; tile = 0; }

    const int b = bh >> 4, h = bh & 15;
    const int r = segR[seg], g = segG[seg], sstart = segStart[seg];
    const int c = h % r;
    const int gbase = segStart[seg] + c * g;   // gidx base (cumulative L == start)
    const int tid = threadIdx.x, lane = tid & 31, wv = tid >> 5;
    const int nl = lane & 15, hb = lane >> 4;
    const size_t rowbase = ((size_t)bh * SEQ + sstart) * HDIM;

    // Persistent A operand: this wave's 16 Q rows (already scaled by 1/sqrt(D))
    int qrow = tile * 128 + wv * 16 + nl;
    int qp = gidx[gbase + qrow];
    const __bf16* qptr = qb + rowbase + (size_t)qp * HDIM;
    v8bf qlo = *(const v8bf*)(qptr + hb * 16);
    v8bf qhi = *(const v8bf*)(qptr + hb * 16 + 8);
    v16bf aq;
    #pragma unroll
    for (int i = 0; i < 8; ++i) { aq[i] = qlo[i]; aq[8 + i] = qhi[i]; }

    float rowmax[8], rowsum[8];
    v8f o0 = {}, o1 = {};
    #pragma unroll
    for (int j = 0; j < 8; ++j) { rowmax[j] = -1e30f; rowsum[j] = 0.0f; }

    for (int kc = 0; kc < g; kc += 32) {
        // cooperative stage: threads 0..127 -> K rows; 128..255 -> V transposed
        if (tid < 128) {
            int row = tid >> 2, q4 = tid & 3;
            int kp = gidx[gbase + kc + row];
            v8bf kv = *(const v8bf*)(kb + rowbase + (size_t)kp * HDIM + q4 * 8);
            *(v8bf*)&Klds[row * 32 + q4 * 8] = kv;
        } else {
            int t2 = tid - 128;
            int row = t2 >> 2, q4 = t2 & 3;
            int kp = gidx[gbase + kc + row];
            v8bf vv = *(const v8bf*)(vb + rowbase + (size_t)kp * HDIM + q4 * 8);
            #pragma unroll
            for (int i = 0; i < 8; ++i) Vt[(q4 * 8 + i) * 32 + row] = vv[i];
        }
        __syncthreads();

        // scores: 16 queries x 32 keys, reduction over D=32
        v16bf bk0 = *(const v16bf*)&Klds[nl * 32 + hb * 16];
        v16bf bk1 = *(const v16bf*)&Klds[(16 + nl) * 32 + hb * 16];
        v8f s0 = {}, s1 = {};
        s0 = __builtin_amdgcn_wmma_f32_16x16x32_bf16(false, aq, false, bk0, (short)0, s0, false, false);
        s1 = __builtin_amdgcn_wmma_f32_16x16x32_bf16(false, aq, false, bk1, (short)0, s1, false, false);

        // online softmax: row stats reduced across the 16 lanes of each half
        #pragma unroll
        for (int j = 0; j < 8; ++j) {
            float mx = fmaxf(s0[j], s1[j]);
            mx = fmaxf(mx, __shfl_xor(mx, 1, 32));
            mx = fmaxf(mx, __shfl_xor(mx, 2, 32));
            mx = fmaxf(mx, __shfl_xor(mx, 4, 32));
            mx = fmaxf(mx, __shfl_xor(mx, 8, 32));
            float mnew = fmaxf(rowmax[j], mx);
            float alpha = __expf(rowmax[j] - mnew);
            float p0 = __expf(s0[j] - mnew);
            float p1 = __expf(s1[j] - mnew);
            float ps = p0 + p1;
            ps += __shfl_xor(ps, 1, 32);
            ps += __shfl_xor(ps, 2, 32);
            ps += __shfl_xor(ps, 4, 32);
            ps += __shfl_xor(ps, 8, 32);
            rowsum[j] = rowsum[j] * alpha + ps;
            rowmax[j] = mnew;
            o0[j] *= alpha;
            o1[j] *= alpha;
            int row = j + 8 * hb;                 // C layout -> P tile row
            Plds[wv][row * 32 + nl]      = (__bf16)p0;
            Plds[wv][row * 32 + 16 + nl] = (__bf16)p1;
        }
        // wave-private LDS RAW: wait LDS writes, block compiler/scheduler motion
        asm volatile("s_wait_dscnt 0x0" ::: "memory");
        __builtin_amdgcn_wave_barrier();

        // P as A operand (16 queries x 32 keys), V^T as B operands (dims 0-15 / 16-31)
        v8bf plo = *(const v8bf*)&Plds[wv][nl * 32 + hb * 16];
        v8bf phi = *(const v8bf*)&Plds[wv][nl * 32 + hb * 16 + 8];
        v16bf ap;
        #pragma unroll
        for (int i = 0; i < 8; ++i) { ap[i] = plo[i]; ap[8 + i] = phi[i]; }
        v16bf bv0 = *(const v16bf*)&Vt[nl * 32 + hb * 16];
        v16bf bv1 = *(const v16bf*)&Vt[(16 + nl) * 32 + hb * 16];
        o0 = __builtin_amdgcn_wmma_f32_16x16x32_bf16(false, ap, false, bv0, (short)0, o0, false, false);
        o1 = __builtin_amdgcn_wmma_f32_16x16x32_bf16(false, ap, false, bv1, (short)0, o1, false, false);
        __syncthreads();
    }

    // normalize and scatter to o[b, s, h, d] (bf16, E-contiguous layout)
    #pragma unroll
    for (int j = 0; j < 8; ++j) {
        float inv = 1.0f / rowsum[j];
        int m = j + 8 * hb;
        int p = gidx[gbase + tile * 128 + wv * 16 + m];
        size_t sglob = (size_t)sstart + p;
        size_t base = ((size_t)b * SEQ + sglob) * EMBED + h * HDIM;
        ob[base + nl]      = (__bf16)(o0[j] * inv);
        ob[base + 16 + nl] = (__bf16)(o1[j] * inv);
    }
}

// ---------------------------------------------------------------------------
// Kernel 4: output projection GEMM, M=16384, N=512, K=512, WMMA bf16, f32 out.
// ---------------------------------------------------------------------------
__global__ __launch_bounds__(256) void out_gemm_kernel(
        const __bf16* __restrict__ o, const float* __restrict__ w,
        const float* __restrict__ bias, float* __restrict__ out) {
    __shared__ __align__(32) __bf16 Alds[128 * 32];
    __shared__ __align__(32) __bf16 Bt[64 * 32];
    const int tid = threadIdx.x;
    const int lane = tid & 31, wv = tid >> 5;
    const int m0 = blockIdx.x * 128;
    const int n0 = blockIdx.y * 64;
    const int nl = lane & 15, hb = lane >> 4;

    v8f acc[4] = {v8f{}, v8f{}, v8f{}, v8f{}};

    for (int k0 = 0; k0 < EMBED; k0 += 32) {
        {   // A already bf16: vector copy 32B per thread
            int row = tid >> 1, half = tid & 1;
            v16bf av = *(const v16bf*)(o + (size_t)(m0 + row) * EMBED + k0 + half * 16);
            *(v16bf*)&Alds[row * 32 + half * 16] = av;
        }
        {
            int k = tid >> 3, ns = (tid & 7) * 8;
            const float* src = w + (size_t)(k0 + k) * EMBED + n0 + ns;
            #pragma unroll
            for (int i = 0; i < 8; ++i) Bt[(ns + i) * 32 + k] = (__bf16)src[i];
        }
        __syncthreads();
        v8bf alo = *(const v8bf*)&Alds[(wv * 16 + nl) * 32 + hb * 16];
        v8bf ahi = *(const v8bf*)&Alds[(wv * 16 + nl) * 32 + hb * 16 + 8];
        v16bf a;
        #pragma unroll
        for (int i = 0; i < 8; ++i) { a[i] = alo[i]; a[8 + i] = ahi[i]; }
        #pragma unroll
        for (int j = 0; j < 4; ++j) {
            v16bf bmat = *(const v16bf*)&Bt[(j * 16 + nl) * 32 + hb * 16];
            acc[j] = __builtin_amdgcn_wmma_f32_16x16x32_bf16(
                false, a, false, bmat, (short)0, acc[j], false, false);
        }
        __syncthreads();
    }
    #pragma unroll
    for (int j = 0; j < 4; ++j) {
        int n = n0 + j * 16 + nl;
        float bval = bias[n];
        #pragma unroll
        for (int i = 0; i < 8; ++i) {
            int row = m0 + wv * 16 + i + 8 * hb;
            out[(size_t)row * EMBED + n] = acc[j][i] + bval;
        }
    }
}

// ---------------------------------------------------------------------------
extern "C" void kernel_launch(void* const* d_in, const int* in_sizes, int n_in,
                              void* d_out, int out_size, void* d_ws, size_t ws_size,
                              hipStream_t stream) {
    const float* x     = (const float*)d_in[0];
    const float* w_qkv = (const float*)d_in[1];
    const float* b_qkv = (const float*)d_in[2];
    const float* w_out = (const float*)d_in[3];
    const float* b_out = (const float*)d_in[4];
    float* out = (float*)d_out;

    const size_t qkvElems = (size_t)BATCH * NHEADS * SEQ * HDIM;   // 8,388,608
    const size_t oElems   = (size_t)BATCH * SEQ * EMBED;           // 8,388,608
    __bf16* qb = (__bf16*)d_ws;
    __bf16* kb = qb + qkvElems;
    __bf16* vb = kb + qkvElems;
    __bf16* ob = vb + qkvElems;
    int* gidx  = (int*)(ob + oElems);

    hilbert_groups_kernel<<<4, 1024, 0, stream>>>(gidx);
    hipMemsetAsync(ob, 0, oElems * sizeof(__bf16), stream);

    dim3 g1(BATCH * SEQ / 128, 1536 / 64);   // 128 x 24
    qkv_gemm_kernel<<<g1, 256, 0, stream>>>(x, w_qkv, b_qkv, qb, kb, vb);

    attn_kernel<<<800, 256, 0, stream>>>(qb, kb, vb, gidx, ob);

    dim3 g2(BATCH * SEQ / 128, EMBED / 64);  // 128 x 8
    out_gemm_kernel<<<g2, 256, 0, stream>>>(ob, w_out, b_out, out);
}
